// callPredictor_55121610277276
// MI455X (gfx1250) — matrix operand order repaired and checked by reading, not compiled, not run on hardware
//
#include <hip/hip_runtime.h>
#include <math.h>

// ---------------------------------------------------------------------------
// Problem constants (from the reference)
// ---------------------------------------------------------------------------
#define B_   512
#define TS_  256
#define F_   256
#define H_   384
#define L_   3
#define K_   10
#define O_   128
#define CH_  128
#define NJ   1542            // 4H + 2L (true xo width)
#define XN   1664            // NJ padded to a multiple of 128
#define KREC 640             // F + H  (recurrence GEMM K)
#define KC   3840            // K_ * H_ (conv GEMM K)

typedef __bf16 bf16_t;
typedef __attribute__((ext_vector_type(16))) __bf16 bf16x16;
typedef __attribute__((ext_vector_type(8)))  float  f32x8;

// ---------------------------------------------------------------------------
// WMMA helpers (CDNA5 v_wmma_f32_16x16x32_bf16, wave32; A and B both use contiguous k = 16*(lane/16) + s)
// ---------------------------------------------------------------------------
__device__ __forceinline__ f32x8 wmma_bf16(bf16x16 a, bf16x16 b, f32x8 c) {
  return __builtin_amdgcn_wmma_f32_16x16x32_bf16(
      /*neg_a=*/false, a, /*neg_b=*/false, b,
      /*c_mod=*/(short)0, c, /*reuse_a=*/false, /*reuse_b=*/false);
}

// A-matrix 16x32 bf16 fragment (row-major source, 16-B aligned rows):
// lane l: row M=l&15, K = kh..kh+15, kh=(l>=16?16:0).
__device__ __forceinline__ bf16x16 load_a_frag(const bf16_t* base, int ld) {
  int lane = threadIdx.x & 31;
  const bf16_t* p = base + (lane & 15) * ld + ((lane >> 4) << 4);
  union { uint4 u[2]; bf16x16 v; } cvt;
  cvt.u[0] = *reinterpret_cast<const uint4*>(p);
  cvt.u[1] = *reinterpret_cast<const uint4*>(p + 8);
  return cvt.v;
}

// B-matrix 32x16 fragment from fragment-major (swizzled) storage: each lane's
// 16 bf16 are contiguous (32 bytes) -> 2x b128 loads.
__device__ __forceinline__ bf16x16 load_b_frag_sw(const bf16_t* tileBase) {
  int lane = threadIdx.x & 31;
  const uint4* p = reinterpret_cast<const uint4*>(tileBase + lane * 16);
  union { uint4 u[2]; bf16x16 v; } cvt;
  cvt.u[0] = p[0];
  cvt.u[1] = p[1];
  return cvt.v;
}

// Swizzled index for element (k, n) of a (K x N) B matrix, 32x16 K-tiles.
__device__ __forceinline__ size_t swz_idx(int k, int n, int nTiles) {
  int kt   = k >> 5, nt = n >> 4;
  int lane = (n & 15) | (((k >> 4) & 1) << 4);
  int e    = k & 15;
  return ((size_t)(kt * nTiles + nt) * 32 + lane) * 16 + e;
}

__device__ __forceinline__ float sigmoidf_(float x) {
  return 1.0f / (1.0f + __expf(-x));
}

// ---------------------------------------------------------------------------
// CDNA5 async global->LDS DMA (ASYNCcnt-tracked; ISA sect. 10 / 15.18)
// dst: wave-relative LDS byte offset (low 32 bits of a generic LDS pointer).
// ---------------------------------------------------------------------------
__device__ __forceinline__ void async_copy_b128(unsigned int ldsOff, const void* gsrc) {
  asm volatile("global_load_async_to_lds_b128 %0, %1, off"
               :: "v"(ldsOff), "v"((unsigned long long)(uintptr_t)gsrc)
               : "memory");
}
__device__ __forceinline__ void wait_asynccnt_le4() {
  asm volatile("s_wait_asynccnt 4" ::: "memory");
}
__device__ __forceinline__ void wait_asynccnt_0() {
  asm volatile("s_wait_asynccnt 0" ::: "memory");
}

// ---------------------------------------------------------------------------
// One-time prep kernels
// ---------------------------------------------------------------------------
__global__ void zero_kernel(float* p, int n) {
  int i = blockIdx.x * blockDim.x + threadIdx.x;
  if (i < n) p[i] = 0.0f;
}

__global__ void cvt_kernel(const float* s, bf16_t* d, int n) {
  int i = blockIdx.x * blockDim.x + threadIdx.x;
  if (i < n) d[i] = (bf16_t)s[i];
}

// Row-major (Kd x Nd) f32 -> swizzled bf16.
__global__ void cvt_sw_kernel(const float* src, bf16_t* dst, int Kd, int Nd) {
  int idx = blockIdx.x * blockDim.x + threadIdx.x;
  if (idx >= Kd * Nd) return;
  int k = idx / Nd, n = idx % Nd;
  dst[swz_idx(k, n, Nd >> 4)] = (bf16_t)src[idx];
}

// W12 swizzled: rows 0..255 = W1[f], rows 256..639 = W2[h]; cols padded to XN.
__global__ void pack_w12_sw_kernel(const float* W1, const float* W2, bf16_t* W12) {
  int idx = blockIdx.x * blockDim.x + threadIdx.x;
  if (idx >= KREC * XN) return;
  int k = idx / XN, j = idx % XN;
  float v = 0.0f;
  if (j < NJ) v = (k < F_) ? W1[k * NJ + j] : W2[(k - F_) * NJ + j];
  W12[swz_idx(k, j, XN >> 4)] = (bf16_t)v;
}

// wt[j] = W1[F,j] + W2[H,j] (the "ti" rank-1 row), bsum[j] = b1[j] + b2[j].
__global__ void pack_vec_kernel(const float* W1, const float* W2,
                                const float* b1, const float* b2,
                                float* wt, float* bsum) {
  int j = blockIdx.x * blockDim.x + threadIdx.x;
  if (j >= XN) return;
  wt[j]   = (j < NJ) ? (W1[F_ * NJ + j] + W2[H_ * NJ + j]) : 0.0f;
  bsum[j] = (j < NJ) ? (b1[j] + b2[j]) : 0.0f;
}

// Conv weights swizzled: logical B[kidx][o] = Wc[o][i][k], kidx = k*H + i.
__global__ void pack_wc_sw_kernel(const float* Wc, bf16_t* Wcb) {
  int idx = blockIdx.x * blockDim.x + threadIdx.x;
  if (idx >= KC * H_) return;
  int kidx = idx / H_, o = idx % H_;
  int k = kidx / H_, i = kidx % H_;
  Wcb[swz_idx(kidx, o, H_ >> 4)] = (bf16_t)Wc[((size_t)o * H_ + i) * K_ + k];
}

// ---------------------------------------------------------------------------
// Per-step kernel 1: xo = [x_t | h_prev] @ W12 + ti*wt + bsum
// M=512, N=XN(1664), K=640.  grid(4,13), block 256 (8 waves).
// WG tile 128x128; waves 2(M)x4(N); wave tile 64x32 (4x2 WMMA accums).
// Double-buffered GLOBAL_LOAD_ASYNC_TO_LDS_B128 staging: tile k+1 DMAs into
// LDS while tile k is multiplied; s_wait_asynccnt 4 retires the current tile.
// ---------------------------------------------------------------------------
__global__ void gemm_xo_kernel(const bf16_t* __restrict__ Xbf,
                               const bf16_t* __restrict__ Hbfp,
                               const bf16_t* __restrict__ W12sw,
                               const float* __restrict__ wt,
                               const float* __restrict__ bsum,
                               const float* __restrict__ T,
                               float* __restrict__ xo, int t) {
  __shared__ __align__(16) bf16_t As[2][128 * 32];
  __shared__ __align__(16) bf16_t Bs[2][8 * 512];
  const int mBase  = blockIdx.x * 128;
  const int nBase  = blockIdx.y * 128;
  const int ntBase = nBase >> 4;
  const int tid    = threadIdx.x;
  const int wave   = tid >> 5;
  const int wm     = wave & 1;
  const int wn     = wave >> 1;

  // per-thread staging sources: uint4 chunks idx=tid and idx=tid+256
  const int r0 = tid >> 2, r1 = (tid + 256) >> 2, c4 = (tid & 3) * 8;
  const bf16_t* aX0 = Xbf + ((size_t)(mBase + r0) * TS_ + t) * F_ + c4;
  const bf16_t* aX1 = Xbf + ((size_t)(mBase + r1) * TS_ + t) * F_ + c4;
  const bf16_t* aH0 = Hbfp + (size_t)(mBase + r0) * H_ + c4;
  const bf16_t* aH1 = Hbfp + (size_t)(mBase + r1) * H_ + c4;
  const bf16_t* bW  = W12sw + (size_t)ntBase * 512 + tid * 8;
  const unsigned int asBase = (unsigned int)(uintptr_t)(void*)&As[0][0] + tid * 16;
  const unsigned int bsBase = (unsigned int)(uintptr_t)(void*)&Bs[0][0] + tid * 16;

  auto stage = [&](int kt, int buf) {
    const bf16_t* a0 = (kt < 8) ? (aX0 + kt * 32) : (aH0 + (kt - 8) * 32);
    const bf16_t* a1 = (kt < 8) ? (aX1 + kt * 32) : (aH1 + (kt - 8) * 32);
    const bf16_t* b0 = bW + (size_t)kt * ((XN >> 4) * 512);
    unsigned int aDst = asBase + buf * (128 * 32 * 2);
    unsigned int bDst = bsBase + buf * (8 * 512 * 2);
    async_copy_b128(aDst,        a0);
    async_copy_b128(aDst + 4096, a1);
    async_copy_b128(bDst,        b0);
    async_copy_b128(bDst + 4096, b0 + 2048);
  };

  f32x8 acc[4][2] = {};
  stage(0, 0);
  const int nKT = KREC / 32;     // 20
  for (int kt = 0; kt < nKT; ++kt) {
    const int buf = kt & 1;
    if (kt + 1 < nKT) { stage(kt + 1, buf ^ 1); wait_asynccnt_le4(); }
    else              { wait_asynccnt_0(); }
    __syncthreads();

    bf16x16 bf[2];
#pragma unroll
    for (int ni = 0; ni < 2; ++ni)
      bf[ni] = load_b_frag_sw(&Bs[buf][(wn * 2 + ni) * 512]);
#pragma unroll
    for (int mi = 0; mi < 4; ++mi) {
      bf16x16 a = load_a_frag(&As[buf][(wm * 64 + mi * 16) * 32], 32);
#pragma unroll
      for (int ni = 0; ni < 2; ++ni)
        acc[mi][ni] = wmma_bf16(a, bf[ni], acc[mi][ni]);
    }
    __syncthreads();
  }

  const int lane = tid & 31;
  const int cN   = lane & 15;
  const int rOff = (lane >> 4) << 3;
#pragma unroll
  for (int mi = 0; mi < 4; ++mi)
#pragma unroll
    for (int ni = 0; ni < 2; ++ni) {
      int col = nBase + wn * 32 + ni * 16 + cN;
#pragma unroll
      for (int v = 0; v < 8; ++v) {
        int row  = mBase + wm * 64 + mi * 16 + rOff + v;
        float ti = T[(size_t)row * TS_ + t];
        xo[(size_t)row * XN + col] = acc[mi][ni][v] + ti * wt[col] + bsum[col];
      }
    }
}

// ---------------------------------------------------------------------------
// Per-step kernel 2: cell nonlinearity + K=10 window + local_dis softmax.
// One block (384 threads) per batch row b.
// ---------------------------------------------------------------------------
__global__ void cell_window_kernel(const float* __restrict__ xo,
                                   float* __restrict__ Cst,
                                   float* __restrict__ Hhist,
                                   bf16_t* __restrict__ Hbf,
                                   float* __restrict__ Dhist,
                                   bf16_t* __restrict__ localh,
                                   bf16_t* __restrict__ meanh, int t) {
  const int b = blockIdx.x;
  const int i = threadIdx.x;           // hidden index 0..383
  __shared__ float sfm[L_], sim[L_], sdis[K_];
  const float* xrow = xo + (size_t)b * XN;

  if (i == 0) {
    float m0 = fmaxf(xrow[0], fmaxf(xrow[1], xrow[2]));
    float e0 = __expf(xrow[0] - m0), e1 = __expf(xrow[1] - m0), e2 = __expf(xrow[2] - m0);
    float s  = e0 + e1 + e2;
    float f0 = e0 / s, f1 = e1 / s, f2 = e2 / s;
    sfm[0] = f0; sfm[1] = f0 + f1; sfm[2] = f0 + f1 + f2;
    float m1 = fmaxf(xrow[3], fmaxf(xrow[4], xrow[5]));
    float g0 = __expf(xrow[3] - m1), g1 = __expf(xrow[4] - m1), g2 = __expf(xrow[5] - m1);
    float q  = g0 + g1 + g2;
    float i0 = g0 / q, i1 = g1 / q, i2 = g2 / q;
    sim[0] = i0 + i1 + i2; sim[1] = i1 + i2; sim[2] = i2;
    float cd = 1.0f - (sfm[0] + sfm[1] + sfm[2]) * (1.0f / 3.0f);
    Dhist[(t % K_) * B_ + b] = cd;
  }
  __syncthreads();

  const int l  = i >> 7;
  float fg  = sigmoidf_(xrow[6 +            i]);
  float ig  = sigmoidf_(xrow[6 +  1 * H_ +  i]);
  float og  = sigmoidf_(xrow[6 +  2 * H_ +  i]);
  float cin = tanhf    (xrow[6 +  3 * H_ +  i]);
  float cl  = Cst[(size_t)b * H_ + i];
  float fm = sfm[l], im = sim[l], ov = fm * im;
  float cn = ov * (fg * cl + ig * cin) + (fm - ov) * cl + (im - ov) * cin;
  float hn = og * tanhf(cn);
  Cst[(size_t)b * H_ + i] = cn;
  Hhist[(size_t)(t % K_) * B_ * H_ + (size_t)b * H_ + i] = hn;
  Hbf  [(size_t)(t % K_) * B_ * H_ + (size_t)b * H_ + i] = (bf16_t)hn;
  __syncthreads();

  if (i == 0) {
    float tmp[K_]; float cs = 0.0f;
    for (int k = 0; k < K_; ++k) {
      int tj = t - (K_ - 1) + k;
      float d = (tj < 0) ? 0.0f : Dhist[(tj % K_) * B_ + b];
      cs += d; tmp[k] = cs;
    }
    float mx = tmp[0];
    for (int k = 1; k < K_; ++k) mx = fmaxf(mx, tmp[k]);
    float sum = 0.0f;
    for (int k = 0; k < K_; ++k) { tmp[k] = __expf(tmp[k] - mx); sum += tmp[k]; }
    float inv = 1.0f / sum;
    for (int k = 0; k < K_; ++k) sdis[k] = tmp[k] * inv;
  }
  __syncthreads();

  float accm = 0.0f;
  for (int k = 0; k < K_; ++k) {
    int tj = t - (K_ - 1) + k;
    float hv = (tj < 0) ? 0.0f
                        : Hhist[(size_t)(tj % K_) * B_ * H_ + (size_t)b * H_ + i];
    float lh = hv * sdis[k];
    localh[(size_t)b * KC + k * H_ + i] = (bf16_t)lh;
    accm += lh;
  }
  meanh[(size_t)b * H_ + i] = (bf16_t)(accm * (1.0f / K_));
}

// ---------------------------------------------------------------------------
// Per-step kernel 3: theme = sigmoid(relu(mean @ Ws + bs) @ Wr + br)
// grid 8 (64 rows each), block 256 (8 waves). Phase-1 intermediate in LDS.
// ---------------------------------------------------------------------------
__global__ void theme_kernel(const bf16_t* __restrict__ meanh,
                             const bf16_t* __restrict__ Wssw, const float* __restrict__ bs,
                             const bf16_t* __restrict__ Wrsw, const float* __restrict__ br,
                             float* __restrict__ theme) {
  __shared__ __align__(16) bf16_t mid[64 * 80];
  const int mBase = blockIdx.x * 64;
  const int wave  = threadIdx.x >> 5;
  const int lane  = threadIdx.x & 31;
  const int cN    = lane & 15;
  const int rOff  = (lane >> 4) << 3;

  {
    const int mt  = wave & 3;
    const int ntg = wave >> 2;
    f32x8 acc1[2] = {};
    for (int k0 = 0; k0 < H_; k0 += 32) {
      bf16x16 a = load_a_frag(meanh + (size_t)(mBase + mt * 16) * H_ + k0, H_);
#pragma unroll
      for (int ni = 0; ni < 2; ++ni) {
        bf16x16 b = load_b_frag_sw(Wssw + (size_t)((k0 >> 5) * 4 + ntg * 2 + ni) * 512);
        acc1[ni] = wmma_bf16(a, b, acc1[ni]);
      }
    }
#pragma unroll
    for (int ni = 0; ni < 2; ++ni) {
      int col = ntg * 32 + ni * 16 + cN;
#pragma unroll
      for (int v = 0; v < 8; ++v) {
        int row = mt * 16 + rOff + v;
        mid[row * 80 + col] = (bf16_t)fmaxf(acc1[ni][v] + bs[col], 0.0f);
      }
    }
  }
  __syncthreads();

  {
    const int mt = wave & 3;
    const int ng = wave >> 2;
    f32x8 acc2[12] = {};
    for (int k0 = 0; k0 < 64; k0 += 32) {
      bf16x16 a = load_a_frag(&mid[(mt * 16) * 80 + k0], 80);
#pragma unroll
      for (int ni = 0; ni < 12; ++ni) {
        bf16x16 b = load_b_frag_sw(Wrsw + (size_t)((k0 >> 5) * 24 + ng * 12 + ni) * 512);
        acc2[ni] = wmma_bf16(a, b, acc2[ni]);
      }
    }
#pragma unroll
    for (int ni = 0; ni < 12; ++ni) {
      int col = ng * 192 + ni * 16 + cN;
#pragma unroll
      for (int v = 0; v < 8; ++v) {
        int row = mBase + mt * 16 + rOff + v;
        theme[(size_t)row * H_ + col] = sigmoidf_(acc2[ni][v] + br[col]);
      }
    }
  }
}

// ---------------------------------------------------------------------------
// Per-step kernel 4: conv = local_h @ Wc + bc; Aout = bf16(theme*conv + h_new)
// M=512, N=384, K=3840.  grid(4,3), block 256; async double-buffered staging.
// ---------------------------------------------------------------------------
__global__ void conv_kernel(const bf16_t* __restrict__ localh,
                            const bf16_t* __restrict__ Wcsw,
                            const float* __restrict__ bc,
                            const float* __restrict__ theme,
                            const float* __restrict__ Hcur,
                            bf16_t* __restrict__ Aout) {
  __shared__ __align__(16) bf16_t As[2][128 * 32];
  __shared__ __align__(16) bf16_t Bs[2][8 * 512];
  const int mBase  = blockIdx.x * 128;
  const int nBase  = blockIdx.y * 128;
  const int ntBase = nBase >> 4;
  const int tid    = threadIdx.x;
  const int wave   = tid >> 5;
  const int wm     = wave & 1;
  const int wn     = wave >> 1;

  const int r0 = tid >> 2, r1 = (tid + 256) >> 2, c4 = (tid & 3) * 8;
  const bf16_t* aS0 = localh + (size_t)(mBase + r0) * KC + c4;
  const bf16_t* aS1 = localh + (size_t)(mBase + r1) * KC + c4;
  const bf16_t* bW  = Wcsw + (size_t)ntBase * 512 + tid * 8;
  const unsigned int asBase = (unsigned int)(uintptr_t)(void*)&As[0][0] + tid * 16;
  const unsigned int bsBase = (unsigned int)(uintptr_t)(void*)&Bs[0][0] + tid * 16;

  auto stage = [&](int kt, int buf) {
    unsigned int aDst = asBase + buf * (128 * 32 * 2);
    unsigned int bDst = bsBase + buf * (8 * 512 * 2);
    const bf16_t* b0 = bW + (size_t)kt * ((H_ >> 4) * 512);
    async_copy_b128(aDst,        aS0 + kt * 32);
    async_copy_b128(aDst + 4096, aS1 + kt * 32);
    async_copy_b128(bDst,        b0);
    async_copy_b128(bDst + 4096, b0 + 2048);
  };

  f32x8 acc[4][2] = {};
  stage(0, 0);
  const int nKT = KC / 32;       // 120
  for (int kt = 0; kt < nKT; ++kt) {
    const int buf = kt & 1;
    if (kt + 1 < nKT) { stage(kt + 1, buf ^ 1); wait_asynccnt_le4(); }
    else              { wait_asynccnt_0(); }
    __syncthreads();

    bf16x16 bf[2];
#pragma unroll
    for (int ni = 0; ni < 2; ++ni)
      bf[ni] = load_b_frag_sw(&Bs[buf][(wn * 2 + ni) * 512]);
#pragma unroll
    for (int mi = 0; mi < 4; ++mi) {
      bf16x16 a = load_a_frag(&As[buf][(wm * 64 + mi * 16) * 32], 32);
#pragma unroll
      for (int ni = 0; ni < 2; ++ni)
        acc[mi][ni] = wmma_bf16(a, bf[ni], acc[mi][ni]);
    }
    __syncthreads();
  }

  const int lane = tid & 31;
  const int cN   = lane & 15;
  const int rOff = (lane >> 4) << 3;
#pragma unroll
  for (int mi = 0; mi < 4; ++mi)
#pragma unroll
    for (int ni = 0; ni < 2; ++ni) {
      int col = nBase + wn * 32 + ni * 16 + cN;
#pragma unroll
      for (int v = 0; v < 8; ++v) {
        int row = mBase + wm * 64 + mi * 16 + rOff + v;
        float cv = acc[mi][ni][v] + bc[col];
        float lo = theme[(size_t)row * H_ + col] * cv;
        float hn = Hcur[(size_t)row * H_ + col];
        Aout[(size_t)row * H_ + col] = (bf16_t)(lo + hn);
      }
    }
}

// ---------------------------------------------------------------------------
// Per-step kernel 5: all_output[:,t,:] = sigmoid(Aout @ Wo + bo); accumulate
// cur_output += all_output * cur_M[:,t].  M=512, N=128, K=384. grid 4.
// ---------------------------------------------------------------------------
__global__ void out_kernel(const bf16_t* __restrict__ Aout,
                           const bf16_t* __restrict__ Wosw,
                           const float* __restrict__ bo,
                           const float* __restrict__ curM,
                           float* __restrict__ out,
                           float* __restrict__ curacc, int t) {
  const int mBase = blockIdx.x * 128;
  const int wave  = threadIdx.x >> 5;
  const int wm    = wave & 1;
  const int wn    = wave >> 1;

  f32x8 acc[4][2] = {};
  for (int k0 = 0; k0 < H_; k0 += 32) {
    bf16x16 bf[2];
#pragma unroll
    for (int ni = 0; ni < 2; ++ni)
      bf[ni] = load_b_frag_sw(Wosw + (size_t)((k0 >> 5) * 8 + wn * 2 + ni) * 512);
#pragma unroll
    for (int mi = 0; mi < 4; ++mi) {
      bf16x16 a = load_a_frag(Aout + (size_t)(mBase + wm * 64 + mi * 16) * H_ + k0, H_);
#pragma unroll
      for (int ni = 0; ni < 2; ++ni)
        acc[mi][ni] = wmma_bf16(a, bf[ni], acc[mi][ni]);
    }
  }

  const int lane = threadIdx.x & 31;
  const int cN   = lane & 15;
  const int rOff = (lane >> 4) << 3;
#pragma unroll
  for (int mi = 0; mi < 4; ++mi)
#pragma unroll
    for (int ni = 0; ni < 2; ++ni) {
      int col = wn * 32 + ni * 16 + cN;
#pragma unroll
      for (int v = 0; v < 8; ++v) {
        int row = mBase + wm * 64 + mi * 16 + rOff + v;
        float val = sigmoidf_(acc[mi][ni][v] + bo[col]);
        out[((size_t)row * TS_ + t) * O_ + col] = val;
        curacc[(size_t)row * O_ + col] += val * curM[(size_t)row * TS_ + t];
      }
    }
}

__global__ void copy_acc_kernel(const float* __restrict__ acc, float* __restrict__ out) {
  int i = blockIdx.x * blockDim.x + threadIdx.x;
  if (i < B_ * O_) out[(size_t)B_ * TS_ * O_ + i] = acc[i];
}

// ---------------------------------------------------------------------------
// Host launch
// ---------------------------------------------------------------------------
extern "C" void kernel_launch(void* const* d_in, const int* in_sizes, int n_in,
                              void* d_out, int out_size, void* d_ws, size_t ws_size,
                              hipStream_t stream) {
  const float* X    = (const float*)d_in[0];
  // d_in[1] = M (all ones, unused by the math)
  const float* curM = (const float*)d_in[2];
  const float* T    = (const float*)d_in[3];
  const float* W1   = (const float*)d_in[4];
  const float* b1   = (const float*)d_in[5];
  const float* W2   = (const float*)d_in[6];
  const float* b2   = (const float*)d_in[7];
  const float* Ws   = (const float*)d_in[8];
  const float* bs   = (const float*)d_in[9];
  const float* Wr   = (const float*)d_in[10];
  const float* br   = (const float*)d_in[11];
  const float* Wc   = (const float*)d_in[12];
  const float* bc   = (const float*)d_in[13];
  const float* Wo   = (const float*)d_in[14];
  const float* bo   = (const float*)d_in[15];
  float* out = (float*)d_out;
  char*  wsb = (char*)d_ws;
  (void)in_sizes; (void)n_in; (void)out_size; (void)ws_size;

  size_t off = 0;
  auto alloc = [&](size_t bytes) {
    size_t o = off; off = (off + bytes + 255) & ~(size_t)255; return o;
  };
  float*  xo     = (float*) (wsb + alloc((size_t)B_ * XN * 4));
  float*  Hhist  = (float*) (wsb + alloc((size_t)K_ * B_ * H_ * 4));
  bf16_t* Hbf    = (bf16_t*)(wsb + alloc((size_t)K_ * B_ * H_ * 2));
  float*  Dhist  = (float*) (wsb + alloc((size_t)K_ * B_ * 4));
  float*  Cst    = (float*) (wsb + alloc((size_t)B_ * H_ * 4));
  float*  curacc = (float*) (wsb + alloc((size_t)B_ * O_ * 4));
  float*  theme  = (float*) (wsb + alloc((size_t)B_ * H_ * 4));
  bf16_t* localh = (bf16_t*)(wsb + alloc((size_t)B_ * KC * 2));
  bf16_t* meanh  = (bf16_t*)(wsb + alloc((size_t)B_ * H_ * 2));
  bf16_t* Aout   = (bf16_t*)(wsb + alloc((size_t)B_ * H_ * 2));
  bf16_t* Xbf    = (bf16_t*)(wsb + alloc((size_t)B_ * TS_ * F_ * 2));
  bf16_t* W12sw  = (bf16_t*)(wsb + alloc((size_t)KREC * XN * 2));
  float*  wt     = (float*) (wsb + alloc((size_t)XN * 4));
  float*  bsum   = (float*) (wsb + alloc((size_t)XN * 4));
  bf16_t* Wcsw   = (bf16_t*)(wsb + alloc((size_t)KC * H_ * 2));
  bf16_t* Wssw   = (bf16_t*)(wsb + alloc((size_t)H_ * 64 * 2));
  bf16_t* Wrsw   = (bf16_t*)(wsb + alloc((size_t)64 * H_ * 2));
  bf16_t* Wosw   = (bf16_t*)(wsb + alloc((size_t)H_ * O_ * 2));

  // ---- deterministic state init (every call)
  {
    int n;
    n = K_ * B_ * H_;      zero_kernel<<<(n + 255) / 256, 256, 0, stream>>>(Hhist, n);
    n = K_ * B_ * H_ / 2;  zero_kernel<<<(n + 255) / 256, 256, 0, stream>>>((float*)Hbf, n);
    n = K_ * B_;           zero_kernel<<<(n + 255) / 256, 256, 0, stream>>>(Dhist, n);
    n = B_ * H_;           zero_kernel<<<(n + 255) / 256, 256, 0, stream>>>(Cst, n);
    n = B_ * O_;           zero_kernel<<<(n + 255) / 256, 256, 0, stream>>>(curacc, n);
  }
  // ---- activations/weights -> bf16 (weights in fragment-major layout)
  cvt_kernel<<<((int)((size_t)B_ * TS_ * F_) + 255) / 256, 256, 0, stream>>>(
      X, Xbf, B_ * TS_ * F_);
  pack_w12_sw_kernel<<<(KREC * XN + 255) / 256, 256, 0, stream>>>(W1, W2, W12sw);
  pack_vec_kernel<<<(XN + 255) / 256, 256, 0, stream>>>(W1, W2, b1, b2, wt, bsum);
  pack_wc_sw_kernel<<<(KC * H_ + 255) / 256, 256, 0, stream>>>(Wc, Wcsw);
  cvt_sw_kernel<<<(H_ * 64 + 255) / 256, 256, 0, stream>>>(Ws, Wssw, H_, 64);
  cvt_sw_kernel<<<(64 * H_ + 255) / 256, 256, 0, stream>>>(Wr, Wrsw, 64, H_);
  cvt_sw_kernel<<<(H_ * O_ + 255) / 256, 256, 0, stream>>>(Wo, Wosw, H_, O_);

  // ---- sequential scan over timesteps
  for (int t = 0; t < TS_; ++t) {
    const bf16_t* Hbfp = Hbf   + (size_t)((t + K_ - 1) % K_) * B_ * H_;
    const float*  Hcur = Hhist + (size_t)(t % K_) * B_ * H_;

    gemm_xo_kernel<<<dim3(B_ / 128, XN / 128), 256, 0, stream>>>(
        Xbf, Hbfp, W12sw, wt, bsum, T, xo, t);
    cell_window_kernel<<<B_, H_, 0, stream>>>(
        xo, Cst, Hhist, Hbf, Dhist, localh, meanh, t);
    theme_kernel<<<B_ / 64, 256, 0, stream>>>(
        meanh, Wssw, bs, Wrsw, br, theme);
    conv_kernel<<<dim3(B_ / 128, H_ / 128), 256, 0, stream>>>(
        localh, Wcsw, bc, theme, Hcur, Aout);
    out_kernel<<<B_ / 128, 256, 0, stream>>>(
        Aout, Wosw, bo, curM, out, curacc, t);
  }

  copy_acc_kernel<<<(B_ * O_ + 255) / 256, 256, 0, stream>>>(curacc, out);
}
